// SharedSirenEncoder_12429635354817
// MI455X (gfx1250) — matrix operand order repaired and checked by reading, not compiled, not run on hardware
//
#include <hip/hip_runtime.h>
#include <hip/hip_bf16.h>
#include <math.h>

typedef __attribute__((ext_vector_type(16))) _Float16 v16h;
typedef __attribute__((ext_vector_type(8)))  _Float16 v8h;
typedef __attribute__((ext_vector_type(8)))  float    v8f;
typedef __attribute__((ext_vector_type(4)))  float    v4f;   // clang vector: OK for NT builtins

#define OMEGA0 30.0f

// ---- packed fp16 weight layout inside d_ws (offsets in halfs) ----
// s1_W [256][128] @ 0        (32768)
// s2_W [512][256] @ 32768    (131072)
// fc1_W[128][512] @ 163840   (65536)
// fc2_W[128][128] @ 229376   (16384)
// fc3_W[512][128] @ 245760   (65536)   total = 311296 halfs = 622592 B

__device__ __forceinline__ v16h cat8(v8h lo, v8h hi) {
    return __builtin_shufflevector(lo, hi, 0,1,2,3,4,5,6,7,8,9,10,11,12,13,14,15);
}

// A fragment (16x32 f16, rows m0..m0+15, k-tile kt) from row-major LDS [128][K].
// Lanes 0-15 hold K={0..7,16..23}, lanes 16-31 hold K={8..15,24..31} (ISA 7.12.2).
template<int K>
__device__ __forceinline__ v16h load_a(const _Float16* lds, int m0, int lm, int hf, int kt) {
    const _Float16* p = lds + (m0 + lm) * K + kt * 32 + hf * 8;
    v8h lo = *(const v8h*)p;
    v8h hi = *(const v8h*)(p + 16);
    return cat8(lo, hi);
}

// B fragment (32x16 f16, cols n0..n0+15, k-tile kt) from row-major weights [O][K]
// (K contiguous per output => exactly the column-major-B layout WMMA wants).
// Lanes 0-15 hold K={0..7,16..23}, lanes 16-31 hold K={8..15,24..31} of their column.
template<int K>
__device__ __forceinline__ v16h load_b(const _Float16* gW, int n0, int lm, int hf, int kt) {
    const _Float16* p = gW + (size_t)(n0 + lm) * K + kt * 32 + hf * 8;
    v8h lo = *(const v8h*)p;
    v8h hi = *(const v8h*)(p + 16);
    return cat8(lo, hi);
}

// Generic fused GEMM + activation. ACT: 0 = sin(omega*x), 1 = relu.
// Wave handles rows [m0, m0+16); iterates all O/16 column tiles.
// Wave-private rows => no cross-wave LDS deps, no barrier needed around this.
template<int K, int O, int ACT>
__device__ __forceinline__ void gemm_layer(const _Float16* __restrict__ ldsIn,
                                           _Float16* __restrict__ ldsOut,
                                           const _Float16* __restrict__ gW,
                                           const float* __restrict__ gB,
                                           int m0, int lane) {
    const int hf = lane >> 4, lm = lane & 15;
    constexpr int KT = K / 32;
    constexpr bool CACHE_A = (KT <= 8);       // <=64 VGPRs of A cached across col tiles
    __builtin_prefetch(gW, 0, 2);             // global_prefetch_b8: warm weight panel
    v16h afrag[CACHE_A ? KT : 1];
    if constexpr (CACHE_A) {
#pragma unroll
        for (int kt = 0; kt < KT; ++kt) afrag[kt] = load_a<K>(ldsIn, m0, lm, hf, kt);
    }
    for (int n0 = 0; n0 < O; n0 += 16) {
        const float bias = gB[n0 + lm];
        v8f acc = {bias, bias, bias, bias, bias, bias, bias, bias};
#pragma unroll
        for (int kt = 0; kt < KT; ++kt) {
            v16h a;
            if constexpr (CACHE_A) a = afrag[kt];
            else                   a = load_a<K>(ldsIn, m0, lm, hf, kt);
            const v16h b = load_b<K>(gW, n0, lm, hf, kt);
            acc = __builtin_amdgcn_wmma_f32_16x16x32_f16(false, a, false, b,
                                                         (short)0, acc, false, false);
        }
        const int rbase = m0 + (hf << 3);     // D: row = rbase+v, col = n0+lm
#pragma unroll
        for (int v = 0; v < 8; ++v) {
            float val = acc[v];
            val = (ACT == 0) ? __sinf(OMEGA0 * val) : fmaxf(val, 0.0f);
            ldsOut[(rbase + v) * O + n0 + lm] = (_Float16)val;
        }
    }
}

// fc3: GEMM + bias + residual(h) + relu, fp32 NT store to global (write-once,
// keep the 1 GiB output stream from evicting L2-resident weight panels).
__device__ __forceinline__ void gemm_out(const _Float16* __restrict__ ldsIn /*[128][128]*/,
                                         const _Float16* __restrict__ ldsH  /*[128][512]*/,
                                         const _Float16* __restrict__ gW    /*[512][128]*/,
                                         const float* __restrict__ gB,
                                         float* __restrict__ out, size_t row0,
                                         int m0, int lane) {
    constexpr int K = 128, O = 512, KT = K / 32;
    const int hf = lane >> 4, lm = lane & 15;
    v16h afrag[KT];
#pragma unroll
    for (int kt = 0; kt < KT; ++kt) afrag[kt] = load_a<K>(ldsIn, m0, lm, hf, kt);
    for (int n0 = 0; n0 < O; n0 += 16) {
        const float bias = gB[n0 + lm];
        v8f acc = {bias, bias, bias, bias, bias, bias, bias, bias};
#pragma unroll
        for (int kt = 0; kt < KT; ++kt) {
            const v16h b = load_b<K>(gW, n0, lm, hf, kt);
            acc = __builtin_amdgcn_wmma_f32_16x16x32_f16(false, afrag[kt], false, b,
                                                         (short)0, acc, false, false);
        }
        const int rbase = m0 + (hf << 3);
#pragma unroll
        for (int v = 0; v < 8; ++v) {
            const float h = (float)ldsH[(rbase + v) * O + n0 + lm];
            const float val = fmaxf(acc[v] + h, 0.0f);
            // half-wave writes 16 contiguous fp32 per row-slice -> coalesced 64B
            __builtin_nontemporal_store(val, out + (row0 + rbase + v) * (size_t)O + n0 + lm);
        }
    }
}

__global__ void convert_weights(const float* __restrict__ s1, const float* __restrict__ s2,
                                const float* __restrict__ f1, const float* __restrict__ f2,
                                const float* __restrict__ f3, _Float16* __restrict__ ws) {
    const int i = blockIdx.x * 256 + threadIdx.x;
    float v;
    if      (i < 32768)  v = s1[i];
    else if (i < 163840) v = s2[i - 32768];
    else if (i < 229376) v = f1[i - 163840];
    else if (i < 245760) v = f2[i - 229376];
    else if (i < 311296) v = f3[i - 245760];
    else return;
    ws[i] = (_Float16)v;
}

__global__ __launch_bounds__(256)
void siren_fused(const float* __restrict__ x,
                 const float* __restrict__ peW, const float* __restrict__ peB,
                 const float* __restrict__ s1B, const float* __restrict__ s2B,
                 const float* __restrict__ f1B, const float* __restrict__ f2B,
                 const float* __restrict__ f3B,
                 const _Float16* __restrict__ wh,
                 float* __restrict__ out) {
    __shared__ alignas(16) _Float16 bufA[128 * 128];   // 32 KB
    __shared__ alignas(16) _Float16 bufB[128 * 256];   // 64 KB
    __shared__ alignas(16) _Float16 bufH[128 * 512];   // 128 KB (kept for residual)
    __shared__ alignas(16) float    xt[128 * 4];       // 2 KB   (226 KB total < 320 KB/WGP)

    const int tid  = threadIdx.x;
    const int lane = tid & 31;
    const int m0   = (tid >> 5) << 4;                  // wave's 16-row slice
    const size_t row0 = (size_t)blockIdx.x * 128;

    // Stage x tile [128][4] into LDS (read-once: NT load, don't pollute caches).
    if (tid < 128) ((v4f*)xt)[tid] =
        __builtin_nontemporal_load((const v4f*)x + row0 + tid);
    __syncthreads();

    // PE layer (K=4: VALU, not WMMA): u = x@peW^T + peB ; h0 = [sin u | cos u] -> bufA[128][128]
    for (int idx = tid; idx < 128 * 64; idx += 256) {
        const int m = idx >> 6, o = idx & 63;
        const v4f xv = ((const v4f*)xt)[m];
        const v4f wv = ((const v4f*)peW)[o];
        const float u = peB[o] + xv.x * wv.x + xv.y * wv.y + xv.z * wv.z + xv.w * wv.w;
        bufA[m * 128 + o]      = (_Float16)__sinf(u);
        bufA[m * 128 + 64 + o] = (_Float16)__cosf(u);
    }
    // PE scattered rows across all threads -> one barrier; afterwards every wave
    // touches only its own 16 rows of every buffer (same-wave LDS is in-order,
    // DScnt), so no further workgroup barriers are needed.
    __syncthreads();

    const _Float16* s1W = wh;
    const _Float16* s2W = wh + 32768;
    const _Float16* f1W = wh + 163840;
    const _Float16* f2W = wh + 229376;
    const _Float16* f3W = wh + 245760;

    gemm_layer<128, 256, 0>(bufA, bufB, s1W, s1B, m0, lane);   // sin(30.)
    gemm_layer<256, 512, 0>(bufB, bufH, s2W, s2B, m0, lane);   // sin(30.) -> h (kept)
    gemm_layer<512, 128, 1>(bufH, bufA, f1W, f1B, m0, lane);   // relu
    gemm_layer<128, 128, 1>(bufA, bufB, f2W, f2B, m0, lane);   // relu
    gemm_out(bufB, bufH, f3W, f3B, out, row0, m0, lane);       // + h, relu, NT store
}

extern "C" void kernel_launch(void* const* d_in, const int* in_sizes, int n_in,
                              void* d_out, int out_size, void* d_ws, size_t ws_size,
                              hipStream_t stream) {
    const float* x   = (const float*)d_in[0];
    const float* peW = (const float*)d_in[1];
    const float* peB = (const float*)d_in[2];
    const float* s1W = (const float*)d_in[3];
    const float* s1B = (const float*)d_in[4];
    const float* s2W = (const float*)d_in[5];
    const float* s2B = (const float*)d_in[6];
    const float* f1W = (const float*)d_in[7];
    const float* f1B = (const float*)d_in[8];
    const float* f2W = (const float*)d_in[9];
    const float* f2B = (const float*)d_in[10];
    const float* f3W = (const float*)d_in[11];
    const float* f3B = (const float*)d_in[12];
    float*     out = (float*)d_out;
    _Float16*  wh  = (_Float16*)d_ws;

    const int N = in_sizes[0] / 4;           // rows (524288)

    convert_weights<<<(311296 + 255) / 256, 256, 0, stream>>>(s1W, s2W, f1W, f2W, f3W, wh);
    siren_fused<<<N / 128, 256, 0, stream>>>(x, peW, peB, s1B, s2B, f1B, f2B, f3B, wh, out);
}
